// RGCNClassifierConcat_88648124991021
// MI455X (gfx1250) — matrix operand order repaired and checked by reading, not compiled, not run on hardware
//
#include <hip/hip_runtime.h>
#include <hip/hip_bf16.h>

#define N_NODES  50000
#define N_EDGES  800000
#define N_RELS   3
#define N_GRAPHS 500
#define EMB      128
#define IN_DIM   384   // 3*EMB
#define HID      256
#define N_CLASS  10

typedef __attribute__((ext_vector_type(16))) _Float16 v16h;
typedef __attribute__((ext_vector_type(8)))  float    v8f;

// ---------------------------------------------------------------------------
// h1[n, 0:384] = [shape_emb[s[n]] | color_emb[c[n]] | pos_emb[p[n]]]
// ---------------------------------------------------------------------------
__global__ __launch_bounds__(256) void embed_kernel(
    const int* __restrict__ s, const int* __restrict__ c, const int* __restrict__ p,
    const float* __restrict__ se, const float* __restrict__ ce, const float* __restrict__ pe,
    float* __restrict__ h1)
{
    int idx = blockIdx.x * 256 + threadIdx.x;
    if (idx >= N_NODES * IN_DIM) return;
    int n = idx / IN_DIM, d = idx - n * IN_DIM;
    float v;
    if (d < EMB)            v = se[(size_t)s[n] * EMB + d];
    else if (d < 2 * EMB)   v = ce[(size_t)c[n] * EMB + (d - EMB)];
    else                    v = pe[(size_t)p[n] * EMB + (d - 2 * EMB)];
    h1[idx] = v;
}

// ---------------------------------------------------------------------------
// cnt[dst*3 + etype] += 1  (f32 atomics; later inverted in place)
// ---------------------------------------------------------------------------
__global__ __launch_bounds__(256) void edge_count_kernel(
    const int* __restrict__ dst, const int* __restrict__ et, float* __restrict__ cnt)
{
    int e = blockIdx.x * 256 + threadIdx.x;
    if (e >= N_EDGES) return;
    atomicAdd(&cnt[dst[e] * N_RELS + et[e]], 1.0f);
}

__global__ __launch_bounds__(256) void invert_cnt_kernel(float* __restrict__ cnt)
{
    int i = blockIdx.x * 256 + threadIdx.x;
    if (i >= N_NODES * N_RELS) return;
    cnt[i] = 1.0f / fmaxf(cnt[i], 1.0f);
}

// ---------------------------------------------------------------------------
// agg[(dst*3+etype)*D + d] += h[src*D + d]   — one block per edge, coalesced
// ---------------------------------------------------------------------------
__global__ __launch_bounds__(256) void scatter_kernel(
    const int* __restrict__ src, const int* __restrict__ dst, const int* __restrict__ et,
    const float* __restrict__ h, float* __restrict__ agg, int D)
{
    int e = blockIdx.x;
    if (e >= N_EDGES) return;
    size_t base = ((size_t)dst[e] * N_RELS + et[e]) * (size_t)D;
    const float* hs = h + (size_t)src[e] * (size_t)D;
    for (int d = threadIdx.x; d < D; d += 256)
        atomicAdd(&agg[base + d], hs[d]);
}

// ---------------------------------------------------------------------------
// WMMA GEMM:  out[N, H] = relu( X[N, 4D] @ Wcat[4D, H] + b )
//   X row n = [ agg[n,0,:]/c0 | agg[n,1,:]/c1 | agg[n,2,:]/c2 | h[n,:] ]
//   Wcat    = [ W (3D x H, contiguous) ; root (D x H) ]
// f32 fidelity via hi/lo f16 split: AB ~= Ah*Bh + Ah*Bl + Al*Bh.
// Block: 8 waves, 128 rows x 64 cols. The 32x64 B tile is split into f16
// hi/lo ONCE at LDS-staging time; waves fetch fragments as ds_load_b128
// (col stride 40 halves = 80B -> every fragment base is 16B-aligned;
// 20-dword lane stride is bank-conflict-free within a 16-lane half).
// ---------------------------------------------------------------------------
__global__ __launch_bounds__(256) void rgcn_gemm_wmma(
    const float* __restrict__ agg, const float* __restrict__ invcnt,
    const float* __restrict__ h,   const float* __restrict__ W,
    const float* __restrict__ root, const float* __restrict__ bias,
    float* __restrict__ out, int N, int D, int H)
{
    const int K3 = 3 * D, Ktot = 4 * D;
    __shared__ __align__(16) _Float16 ldsBhi[64][40];   // [col][K], 80B row stride
    __shared__ __align__(16) _Float16 ldsBlo[64][40];

    const int lane    = threadIdx.x & 31;
    const int wave    = threadIdx.x >> 5;
    const int l15     = lane & 15;
    const int halfSel = lane >> 4;              // 0: lanes 0-15, 1: lanes 16-31

    int rowBase = blockIdx.x * 128 + wave * 16; // wave-uniform
    if (rowBase > N - 16) rowBase = N - 16;     // clamp (N % 16 == 0), keeps EXEC full
    const int colBase = blockIdx.y * 64;
    const int row = rowBase + l15;              // this lane's A row

    v8f acc[4];
    #pragma unroll
    for (int ct = 0; ct < 4; ++ct)
        #pragma unroll
        for (int i = 0; i < 8; ++i) acc[ct][i] = 0.0f;

    union V16U { v16h v; unsigned u[8]; };

    for (int k0 = 0; k0 < Ktot; k0 += 32) {
        __syncthreads();
        {   // stage 32x64 B tile: global f32 -> hi/lo f16 in LDS (each elem once)
            int r  = threadIdx.x & 31;           // K row within tile
            int cg = (threadIdx.x >> 5) * 8;     // 8 columns per thread
            int kg = k0 + r;
            const float* wrow = (kg < K3) ? (W + (size_t)kg * H)
                                          : (root + (size_t)(kg - K3) * H);
            const float4* wv = (const float4*)(wrow + colBase + cg);
            float4 a0 = wv[0], a1 = wv[1];
            float f[8] = { a0.x, a0.y, a0.z, a0.w, a1.x, a1.y, a1.z, a1.w };
            #pragma unroll
            for (int j = 0; j < 8; ++j) {
                _Float16 hi = (_Float16)f[j];
                ldsBhi[cg + j][r] = hi;
                ldsBlo[cg + j][r] = (_Float16)(f[j] - (float)hi);
            }
        }
        __syncthreads();

        // --- build A fragment (hi/lo) : elems 0-7 -> K khalf..+7, 8-15 -> +16 ---
        v16h ahi, alo;
        #pragma unroll
        for (int g = 0; g < 2; ++g) {
            const int kg = k0 + halfSel * 8 + g * 16;   // 8-aligned group start
            float xv[8];
            if (kg < K3) {
                int rel  = (kg >= D) + (kg >= 2 * D);   // uniform within 8-group
                float ic = invcnt[row * N_RELS + rel];
                const float4* ap = (const float4*)(agg + (size_t)row * K3 + kg);
                float4 x0 = ap[0], x1 = ap[1];
                xv[0] = x0.x * ic; xv[1] = x0.y * ic; xv[2] = x0.z * ic; xv[3] = x0.w * ic;
                xv[4] = x1.x * ic; xv[5] = x1.y * ic; xv[6] = x1.z * ic; xv[7] = x1.w * ic;
            } else {
                const float4* hp = (const float4*)(h + (size_t)row * D + (kg - K3));
                float4 x0 = hp[0], x1 = hp[1];
                xv[0] = x0.x; xv[1] = x0.y; xv[2] = x0.z; xv[3] = x0.w;
                xv[4] = x1.x; xv[5] = x1.y; xv[6] = x1.z; xv[7] = x1.w;
            }
            #pragma unroll
            for (int j = 0; j < 8; ++j) {
                _Float16 hi = (_Float16)xv[j];
                ahi[g * 8 + j] = hi;
                alo[g * 8 + j] = (_Float16)(xv[j] - (float)hi);
            }
        }

        // --- 4 column tiles: B fragments = 2x ds_load_b128 each (hi and lo) ---
        const int kb = halfSel * 8;    // lanes 0-15: K 0-7,16-23; lanes 16-31: K 8-15,24-31
        #pragma unroll
        for (int ct = 0; ct < 4; ++ct) {
            const int col = ct * 16 + l15;
            V16U bh, bl;
            __builtin_memcpy(&bh.u[0], &ldsBhi[col][kb],     16);
            __builtin_memcpy(&bh.u[4], &ldsBhi[col][kb + 16], 16);
            __builtin_memcpy(&bl.u[0], &ldsBlo[col][kb],     16);
            __builtin_memcpy(&bl.u[4], &ldsBlo[col][kb + 16], 16);
            acc[ct] = __builtin_amdgcn_wmma_f32_16x16x32_f16(
                false, ahi, false, bh.v, (short)0, acc[ct], false, false);
            acc[ct] = __builtin_amdgcn_wmma_f32_16x16x32_f16(
                false, ahi, false, bl.v, (short)0, acc[ct], false, false);
            acc[ct] = __builtin_amdgcn_wmma_f32_16x16x32_f16(
                false, alo, false, bh.v, (short)0, acc[ct], false, false);
        }
    }

    // epilogue: C/D layout — VGPR i: lanes0-15 (M=i, N=lane), lanes16-31 (M=8+i)
    #pragma unroll
    for (int ct = 0; ct < 4; ++ct) {
        const int col = colBase + ct * 16 + l15;
        const float bv = bias[col];
        #pragma unroll
        for (int i = 0; i < 8; ++i) {
            int m = halfSel * 8 + i;
            float v = acc[ct][i] + bv;
            out[(size_t)(rowBase + m) * H + col] = v > 0.0f ? v : 0.0f;
        }
    }
}

// ---------------------------------------------------------------------------
// Pooling + classifier
// ---------------------------------------------------------------------------
__global__ __launch_bounds__(256) void pool_sum_kernel(
    const int* __restrict__ batch, const float* __restrict__ h, float* __restrict__ pooled)
{
    int n = blockIdx.x;
    if (n >= N_NODES) return;
    int g = batch[n];
    int d = threadIdx.x;                      // HID == 256 == blockDim
    atomicAdd(&pooled[(size_t)g * HID + d], h[(size_t)n * HID + d]);
}

__global__ __launch_bounds__(256) void pool_cnt_kernel(
    const int* __restrict__ batch, float* __restrict__ gcnt)
{
    int n = blockIdx.x * 256 + threadIdx.x;
    if (n >= N_NODES) return;
    atomicAdd(&gcnt[batch[n]], 1.0f);
}

__global__ __launch_bounds__(256) void final_linear_kernel(
    const float* __restrict__ pooled, const float* __restrict__ gcnt,
    const float* __restrict__ lw, const float* __restrict__ lb, float* __restrict__ out)
{
    int idx = blockIdx.x * 256 + threadIdx.x;
    if (idx >= N_GRAPHS * N_CLASS) return;
    int g = idx / N_CLASS, cls = idx - g * N_CLASS;
    float inv = 1.0f / fmaxf(gcnt[g], 1.0f);
    float s = 0.0f;
    for (int k = 0; k < HID; ++k)
        s += pooled[(size_t)g * HID + k] * lw[k * N_CLASS + cls];
    out[idx] = s * inv + lb[cls];
}

// ---------------------------------------------------------------------------
extern "C" void kernel_launch(void* const* d_in, const int* in_sizes, int n_in,
                              void* d_out, int out_size, void* d_ws, size_t ws_size,
                              hipStream_t stream)
{
    const int*   s      = (const int*)  d_in[0];
    const int*   c      = (const int*)  d_in[1];
    const int*   p      = (const int*)  d_in[2];
    const int*   eidx   = (const int*)  d_in[3];   // (2, E): src row 0, dst row 1
    const int*   etype  = (const int*)  d_in[4];
    const int*   batch  = (const int*)  d_in[5];
    const float* se     = (const float*)d_in[6];
    const float* ce     = (const float*)d_in[7];
    const float* pe     = (const float*)d_in[8];
    const float* W1     = (const float*)d_in[9];   // (3,384,256) == (1152,256)
    const float* root1  = (const float*)d_in[10];
    const float* b1     = (const float*)d_in[11];
    const float* W2     = (const float*)d_in[12];  // (3,256,256) == (768,256)
    const float* root2  = (const float*)d_in[13];
    const float* b2     = (const float*)d_in[14];
    const float* lw     = (const float*)d_in[15];
    const float* lb     = (const float*)d_in[16];
    float*       out    = (float*)d_out;

    const int* esrc = eidx;
    const int* edst = eidx + N_EDGES;

    // workspace layout (floats)
    float* ws = (float*)d_ws;
    size_t o = 0;
    float* h1     = ws + o; o += (size_t)N_NODES * IN_DIM;          // 19.2M
    float* agg    = ws + o; o += (size_t)N_NODES * N_RELS * IN_DIM; // 57.6M (reused L2)
    float* invcnt = ws + o; o += (size_t)N_NODES * N_RELS;          // 150K
    float* h2     = ws + o; o += (size_t)N_NODES * HID;             // 12.8M
    float* h3     = ws + o; o += (size_t)N_NODES * HID;             // 12.8M
    float* pooled = ws + o; o += (size_t)N_GRAPHS * HID;
    float* gcnt   = ws + o; o += (size_t)N_GRAPHS;

    // ---- embeddings + edge counts (shared by both layers) ----
    hipMemsetAsync(invcnt, 0, (size_t)N_NODES * N_RELS * sizeof(float), stream);
    embed_kernel<<<(N_NODES * IN_DIM + 255) / 256, 256, 0, stream>>>(s, c, p, se, ce, pe, h1);
    edge_count_kernel<<<(N_EDGES + 255) / 256, 256, 0, stream>>>(edst, etype, invcnt);
    invert_cnt_kernel<<<(N_NODES * N_RELS + 255) / 256, 256, 0, stream>>>(invcnt);

    // ---- layer 1: scatter h1 (D=384), GEMM 50000 x 1536 x 256 -> h2 ----
    hipMemsetAsync(agg, 0, (size_t)N_NODES * N_RELS * IN_DIM * sizeof(float), stream);
    scatter_kernel<<<N_EDGES, 256, 0, stream>>>(esrc, edst, etype, h1, agg, IN_DIM);
    {
        dim3 grid((N_NODES + 127) / 128, HID / 64);
        rgcn_gemm_wmma<<<grid, 256, 0, stream>>>(agg, invcnt, h1, W1, root1, b1,
                                                 h2, N_NODES, IN_DIM, HID);
    }

    // ---- layer 2: scatter h2 (D=256), GEMM 50000 x 1024 x 256 -> h3 ----
    hipMemsetAsync(agg, 0, (size_t)N_NODES * N_RELS * HID * sizeof(float), stream);
    scatter_kernel<<<N_EDGES, 256, 0, stream>>>(esrc, edst, etype, h2, agg, HID);
    {
        dim3 grid((N_NODES + 127) / 128, HID / 64);
        rgcn_gemm_wmma<<<grid, 256, 0, stream>>>(agg, invcnt, h2, W2, root2, b2,
                                                 h3, N_NODES, HID, HID);
    }

    // ---- global mean pool + classifier ----
    hipMemsetAsync(pooled, 0, (size_t)N_GRAPHS * HID * sizeof(float), stream);
    hipMemsetAsync(gcnt,   0, (size_t)N_GRAPHS * sizeof(float), stream);
    pool_sum_kernel<<<N_NODES, 256, 0, stream>>>(batch, h3, pooled);
    pool_cnt_kernel<<<(N_NODES + 255) / 256, 256, 0, stream>>>(batch, gcnt);
    final_linear_kernel<<<(N_GRAPHS * N_CLASS + 255) / 256, 256, 0, stream>>>(
        pooled, gcnt, lw, lb, out);
}
